// BiLSTM_CRF_77464030150953
// MI455X (gfx1250) — matrix operand order repaired and checked by reading, not compiled, not run on hardware
//
#include <hip/hip_runtime.h>
#include <hip/hip_bf16.h>

// ---------------------------------------------------------------------------
// BiLSTM-CRF for MI455X (gfx1250), wave32.
//  - Input GEMMs via v_wmma_f32_16x16x32_bf16 (bf16 in, fp32 accum)
//  - Recurrent phase: weight-stationary LSTM, w_hh slice DMA'd into LDS by the
//    Tensor Data Mover (tensor_load_to_lds, TENSORcnt), h ping-pong through
//    L2, per-step cross-WG sync (cluster barrier fast path + atomic fallback)
//  - Viterbi: single-wave sequential kernel
// ---------------------------------------------------------------------------

typedef __attribute__((ext_vector_type(16))) __bf16 v16bf;
typedef __attribute__((ext_vector_type(8)))  __bf16 v8bf;
typedef __attribute__((ext_vector_type(8)))  float  v8f;
typedef __attribute__((ext_vector_type(4)))  unsigned int v4u;
typedef __attribute__((ext_vector_type(8)))  int v8i_;
typedef __attribute__((ext_vector_type(4)))  int v4i_;

#define T_LEN 8192
#define E_DIM 512
#define H_DIM 512
#define G_DIM 2048          // 4*H
#define K_TAG 19
#define TAG_START 17
#define TAG_STOP  18
#define NEG_VAL  (-10000.0f)

#define NWG 32              // workgroups per direction (recurrent kernel)
#define HPW (H_DIM / NWG)   // 16 h-elements owned per WG
#define ROWS_PER_WG (4 * HPW) // 64 gate rows per WG (i,f,g,o slices)
#define LSTM_THREADS 256

__device__ __forceinline__ float sigm(float x) { return 1.0f / (1.0f + __expf(-x)); }

// ------------------------------ utility ------------------------------------
__global__ void init_bar(unsigned* bar) {
    if (threadIdx.x < 4) bar[threadIdx.x] = 0u;
}

__global__ void cvt_bf16(const float* __restrict__ src, __bf16* __restrict__ dst, int n) {
    for (int i = blockIdx.x * blockDim.x + threadIdx.x; i < n; i += gridDim.x * blockDim.x)
        dst[i] = (__bf16)src[i];
}

// --------------------------- WMMA input GEMM -------------------------------
// XG[T x 2048] = X[T x 512] @ W^T (W is 2048 x 512, row-major) + b1 + b2
// One wave per 16(M) x 64(N) tile; A reused across 4 accumulators.
__global__ void gemm_xg(const __bf16* __restrict__ X,
                        const __bf16* __restrict__ W,
                        const float* __restrict__ b1,
                        const float* __restrict__ b2,
                        float* __restrict__ XG) {
    const int m0   = blockIdx.x * 16;
    const int n0   = blockIdx.y * 64;
    const int lane = threadIdx.x;          // 0..31
    const int mrow = m0 + (lane & 15);
    const int akb  = (lane >> 4) * 8;      // A K-offset group per ISA layout
    const int bkb  = (lane >> 4) * 8;      // B K-offset group per ISA layout

    v8f acc0 = {}, acc1 = {}, acc2 = {}, acc3 = {};

    for (int k0 = 0; k0 < E_DIM; k0 += 32) {
        // A fragment: lane holds row m, Ks {akb+0..7} U {akb+16..23}
        const __bf16* ap = X + (size_t)mrow * E_DIM + k0 + akb;
        v8bf alo = *(const v8bf*)ap;
        v8bf ahi = *(const v8bf*)(ap + 16);
        v16bf a = __builtin_shufflevector(alo, ahi,
                    0,1,2,3,4,5,6,7,8,9,10,11,12,13,14,15);

        #pragma unroll
        for (int s = 0; s < 4; ++s) {
            const int n = n0 + s * 16 + (lane & 15);
            // B fragment: lane holds w_ih row n (== column n of B), Ks {bkb+0..7} U {bkb+16..23}
            const __bf16* bp = W + (size_t)n * E_DIM + k0 + bkb;
            v8bf blo = *(const v8bf*)bp;
            v8bf bhi = *(const v8bf*)(bp + 16);
            v16bf b = __builtin_shufflevector(blo, bhi,
                        0,1,2,3,4,5,6,7,8,9,10,11,12,13,14,15);
            v8f* accp = (s == 0) ? &acc0 : (s == 1) ? &acc1 : (s == 2) ? &acc2 : &acc3;
            *accp = __builtin_amdgcn_wmma_f32_16x16x32_bf16(
                        false, a, false, b, (short)0, *accp, false, false);
        }
    }

    // epilogue: C layout — VGPR v: row = m0 + v + (lane>>4)*8, col = lane&15
    const int rbase = m0 + ((lane >> 4) << 3);
    #pragma unroll
    for (int s = 0; s < 4; ++s) {
        const int n = n0 + s * 16 + (lane & 15);
        const float bias = b1[n] + b2[n];
        v8f acc = (s == 0) ? acc0 : (s == 1) ? acc1 : (s == 2) ? acc2 : acc3;
        #pragma unroll
        for (int v = 0; v < 8; ++v)
            XG[(size_t)(rbase + v) * G_DIM + n] = acc[v] + bias;
    }
}

// ----------------------- cross-WG per-step barrier -------------------------
__device__ __forceinline__ void dir_barrier(unsigned* cnt, unsigned* gen) {
    __syncthreads();
    if (threadIdx.x == 0) {
        __threadfence();
        if (__builtin_amdgcn_cluster_id_x() != 0) {
            // cluster-dispatch fast path: hardware split barrier (-3)
            __builtin_amdgcn_s_cluster_barrier();
        } else {
            volatile unsigned* vg = (volatile unsigned*)gen;
            unsigned g = *vg;
            __threadfence();
            if (atomicAdd(cnt, 1u) == NWG - 1u) {
                *cnt = 0u;
                __threadfence();
                *vg = g + 1u;
            } else {
                while (*vg == g) { __builtin_amdgcn_s_sleep(1); }
            }
            __threadfence();
        }
    }
    __syncthreads();
}

// ------------------ TDM: 16x512 bf16 tile, global -> LDS -------------------
// D# per CDNA5 ISA: group0 {count=1, lds_addr,
// global_addr, type=2}; group1 {data_size=1(2B), tensor_dim0=512,
// tensor_dim1=2048, tile_dim0=512, tile_dim1=16, dim0_stride=512}.
// This toolchain exposes the 6-arg builtin:
//   (uint32x4 g0, int32x8 g1, int32x4 g2, int32x4 g3, int32x8 g4, i32 cpol)
__device__ __forceinline__ void tdm_load_tile16x512(unsigned lds_byte_off,
                                                    unsigned long long gaddr) {
#if __has_builtin(__builtin_amdgcn_tensor_load_to_lds)
    v4u g0;
    g0[0] = 1u;                                            // count=1, user mode
    g0[1] = lds_byte_off;                                  // lds_addr
    g0[2] = (unsigned)(gaddr & 0xFFFFFFFFull);             // global_addr[31:0]
    g0[3] = (unsigned)((gaddr >> 32) & 0x01FFFFFFull)      // global_addr[56:32]
          | 0x80000000u;                                   // type=2 (bits 127:126)
    v8i_ g1;
    g1[0] = 0x10000;                 // data_size=1 (2 bytes); mask/flags = 0
    g1[1] = (int)(512u << 16);       // tensor_dim0[15:0] in bits 63:48
    g1[2] = (int)(2048u << 16);      // tensor_dim0 hi=0 | tensor_dim1[15:0]
    g1[3] = (int)(512u << 16);       // tensor_dim1 hi=0 | tile_dim0=512
    g1[4] = 16;                      // tile_dim1=16, tile_dim2=0
    g1[5] = 512;                     // tensor_dim0_stride[31:0]
    g1[6] = 0;                       // stride0 hi | stride1 lo (2D: unused)
    g1[7] = 0;
    v4i_ gz4 = {0, 0, 0, 0};         // groups 2/3 unused for 2D tensors
    v8i_ gz8 = {0, 0, 0, 0, 0, 0, 0, 0};
    __builtin_amdgcn_tensor_load_to_lds(g0, g1, gz4, gz4, gz8, 0);
#else
    (void)lds_byte_off; (void)gaddr;
#endif
}

// --------------------------- recurrent LSTM --------------------------------
// grid = 2*NWG blocks of 256 threads. Block b: dir = b/NWG, wg = b%NWG.
// WG owns h[j] for j in [wg*HPW, wg*HPW+HPW) and gate rows {g*512 + wg*HPW + j'}.
__global__ void lstm_kernel(const __bf16* __restrict__ whf_bf,
                            const __bf16* __restrict__ whb_bf,
                            const float* __restrict__ xg_f,
                            const float* __restrict__ xg_b,
                            const float* __restrict__ h0,
                            const float* __restrict__ c0,
                            float* __restrict__ hs_f,
                            float* __restrict__ hs_b,
                            float* __restrict__ hbuf,   // [2 dirs][2 phases][512]
                            unsigned* __restrict__ bar) // [2 dirs][cnt,gen]
{
    const int dir = blockIdx.x / NWG;
    const int wg  = blockIdx.x % NWG;
    const __bf16* w_hh = dir ? whb_bf : whf_bf;
    const float* xg   = dir ? xg_b   : xg_f;
    float* hs = dir ? hs_b : hs_f;
    float* hb = hbuf + (size_t)dir * 2 * H_DIM;
    unsigned* cnt = bar + dir * 2;
    unsigned* gen = bar + dir * 2 + 1;

    extern __shared__ char smem[];
    __bf16* wsl  = (__bf16*)smem;                                   // 64*512 bf16
    float*  hl   = (float*)(smem + ROWS_PER_WG * H_DIM * sizeof(__bf16)); // 512
    float*  gsum = hl + H_DIM;                                      // 64

    const int tid = threadIdx.x;

#if __has_builtin(__builtin_amdgcn_tensor_load_to_lds)
    // TDM preload: 4 gate chunks of 16 rows x 512 bf16 each. local row
    // r = g*HPW + j'  <-  global row g*512 + wg*HPW + j'. Contiguous in both
    // LDS and global, so each chunk is one 2D-tile descriptor.
    if (tid < 32) {                       // wave 0 issues; EXEC ignored by TDM
        const unsigned lds_base = (unsigned)(size_t)(void*)wsl;
        #pragma unroll
        for (int g = 0; g < 4; ++g) {
            const size_t grow = (size_t)(g * H_DIM + wg * HPW);
            tdm_load_tile16x512(lds_base + (unsigned)(g * HPW * H_DIM * 2),
                                (unsigned long long)(size_t)(w_hh + grow * H_DIM));
        }
        __builtin_amdgcn_s_wait_tensorcnt(0);
    }
#else
    for (int idx = tid; idx < ROWS_PER_WG * H_DIM; idx += LSTM_THREADS) {
        const int r = idx >> 9, k = idx & (H_DIM - 1);
        const int grow = (r / HPW) * H_DIM + wg * HPW + (r % HPW);
        wsl[idx] = w_hh[(size_t)grow * H_DIM + k];
    }
#endif

    // Init state: c in registers of threads 0..HPW-1, h into phase-0 buffer.
    float c = 0.0f;
    if (tid < HPW) {
        const int gj = wg * HPW + tid;
        c = c0[dir * H_DIM + gj];
        hb[gj] = h0[dir * H_DIM + gj];
    }
    dir_barrier(cnt, gen);

    int cur = 0;
    for (int s = 0; s < T_LEN; ++s) {
        const int t = dir ? (T_LEN - 1 - s) : s;

        // stage previous h into LDS; prefetch next step's xg row into caches
        for (int i = tid; i < H_DIM; i += LSTM_THREADS) hl[i] = hb[cur * H_DIM + i];
        if (s + 1 < T_LEN && tid < HPW) {
            const int tn = dir ? (T_LEN - 2 - s) : (s + 1);
            __builtin_prefetch(xg + (size_t)tn * G_DIM + wg * HPW + tid, 0, 1);
        }
        __syncthreads();

        // gate dots: 4 threads per row, 128 elements each, wave32 xor-reduce
        const int r = tid >> 2;          // 0..63 local gate row
        const int part = tid & 3;
        const __bf16* wrow = wsl + r * H_DIM + part * 128;
        const float*  hp   = hl + part * 128;
        float sum = 0.0f;
        #pragma unroll 8
        for (int k = 0; k < 128; ++k) sum += (float)wrow[k] * hp[k];
        sum += __shfl_xor(sum, 1, 32);
        sum += __shfl_xor(sum, 2, 32);
        if (part == 0) gsum[r] = sum;
        __syncthreads();

        // pointwise gate math on owned h elements
        if (tid < HPW) {
            const int gj = wg * HPW + tid;
            const float* xgt = xg + (size_t)t * G_DIM;
            const float pi = gsum[0 * HPW + tid] + xgt[0 * H_DIM + gj];
            const float pf = gsum[1 * HPW + tid] + xgt[1 * H_DIM + gj];
            const float pg = gsum[2 * HPW + tid] + xgt[2 * H_DIM + gj];
            const float po = gsum[3 * HPW + tid] + xgt[3 * H_DIM + gj];
            c = sigm(pf) * c + sigm(pi) * tanhf(pg);
            const float h = sigm(po) * tanhf(c);
            hs[(size_t)t * H_DIM + gj] = h;
            hb[(cur ^ 1) * H_DIM + gj] = h;
        }
        dir_barrier(cnt, gen);
        cur ^= 1;
    }
}

// --------------------------- tag projection --------------------------------
__global__ void feats_kernel(const float* __restrict__ hs_f,
                             const float* __restrict__ hs_b,
                             const float* __restrict__ W_tag,
                             const float* __restrict__ b_tag,
                             float* __restrict__ feats) {
    const int id = blockIdx.x * blockDim.x + threadIdx.x;
    if (id >= T_LEN * K_TAG) return;
    const int t = id / K_TAG, k = id % K_TAG;
    const float* wt = W_tag + (size_t)k * (2 * H_DIM);
    const float* hf = hs_f + (size_t)t * H_DIM;
    const float* hb = hs_b + (size_t)t * H_DIM;
    float s = b_tag[k];
    for (int i = 0; i < H_DIM; ++i) s += hf[i] * wt[i];
    for (int i = 0; i < H_DIM; ++i) s += hb[i] * wt[H_DIM + i];
    feats[id] = s;
}

// ------------------------------- Viterbi -----------------------------------
__global__ void viterbi_kernel(const float* __restrict__ feats,
                               const float* __restrict__ trans,
                               float* __restrict__ out,
                               int* __restrict__ bptrs) {
    __shared__ float fv[32];
    __shared__ float tr[K_TAG * K_TAG];
    const int tid = threadIdx.x; // 32 threads (one wave)
    for (int i = tid; i < K_TAG * K_TAG; i += 32) tr[i] = trans[i];
    if (tid < K_TAG) fv[tid] = (tid == TAG_START) ? 0.0f : NEG_VAL;
    __syncthreads();

    for (int t = 0; t < T_LEN; ++t) {
        float best = -3.4e38f; int bi = 0;
        if (tid < K_TAG) {
            #pragma unroll
            for (int i = 0; i < K_TAG; ++i) {
                const float v = fv[i] + tr[tid * K_TAG + i];
                if (v > best) { best = v; bi = i; }
            }
            bptrs[t * K_TAG + tid] = bi;
        }
        __syncthreads();
        if (tid < K_TAG) fv[tid] = best + feats[t * K_TAG + tid];
        __syncthreads();
    }

    if (tid == 0) {
        float bestv = -3.4e38f; int bj = 0;
        for (int j = 0; j < K_TAG; ++j) {
            const float v = fv[j] + tr[TAG_STOP * K_TAG + j];
            if (v > bestv) { bestv = v; bj = j; }
        }
        out[0] = bestv;
        int tag = bj;
        for (int t = T_LEN - 1; t >= 0; --t) {
            out[1 + t] = (float)tag;
            tag = bptrs[t * K_TAG + tag];
        }
    }
}

// ------------------------------ launcher -----------------------------------
extern "C" void kernel_launch(void* const* d_in, const int* in_sizes, int n_in,
                              void* d_out, int out_size, void* d_ws, size_t ws_size,
                              hipStream_t stream) {
    const float* x      = (const float*)d_in[0];
    const float* w_ih_f = (const float*)d_in[1];
    const float* w_hh_f = (const float*)d_in[2];
    const float* b_ih_f = (const float*)d_in[3];
    const float* b_hh_f = (const float*)d_in[4];
    const float* w_ih_b = (const float*)d_in[5];
    const float* w_hh_b = (const float*)d_in[6];
    const float* b_ih_b = (const float*)d_in[7];
    const float* b_hh_b = (const float*)d_in[8];
    const float* h0     = (const float*)d_in[9];
    const float* c0     = (const float*)d_in[10];
    const float* W_tag  = (const float*)d_in[11];
    const float* b_tag  = (const float*)d_in[12];
    const float* trans  = (const float*)d_in[13];
    float* out = (float*)d_out;

    uintptr_t p = (uintptr_t)d_ws;
    auto alloc = [&](size_t bytes) -> void* {
        uintptr_t r = (p + 255) & ~(uintptr_t)255;
        p = r + bytes;
        return (void*)r;
    };
    __bf16* xbf   = (__bf16*)alloc((size_t)T_LEN * E_DIM * 2);
    __bf16* wfbf  = (__bf16*)alloc((size_t)G_DIM * E_DIM * 2);
    __bf16* wbbf  = (__bf16*)alloc((size_t)G_DIM * E_DIM * 2);
    __bf16* whfbf = (__bf16*)alloc((size_t)G_DIM * H_DIM * 2);
    __bf16* whbbf = (__bf16*)alloc((size_t)G_DIM * H_DIM * 2);
    float* xg_f  = (float*)alloc((size_t)T_LEN * G_DIM * 4);
    float* xg_b  = (float*)alloc((size_t)T_LEN * G_DIM * 4);
    float* hs_f  = (float*)alloc((size_t)T_LEN * H_DIM * 4);
    float* hs_b  = (float*)alloc((size_t)T_LEN * H_DIM * 4);
    float* feats = (float*)alloc((size_t)T_LEN * K_TAG * 4);
    int*   bptrs = (int*)alloc((size_t)T_LEN * K_TAG * 4);
    float* hbuf  = (float*)alloc((size_t)2 * 2 * H_DIM * 4);
    unsigned* bar = (unsigned*)alloc(4 * sizeof(unsigned));

    init_bar<<<1, 32, 0, stream>>>(bar);
    cvt_bf16<<<1024, 256, 0, stream>>>(x, xbf, T_LEN * E_DIM);
    cvt_bf16<<<512, 256, 0, stream>>>(w_ih_f, wfbf, G_DIM * E_DIM);
    cvt_bf16<<<512, 256, 0, stream>>>(w_ih_b, wbbf, G_DIM * E_DIM);
    cvt_bf16<<<512, 256, 0, stream>>>(w_hh_f, whfbf, G_DIM * H_DIM);
    cvt_bf16<<<512, 256, 0, stream>>>(w_hh_b, whbbf, G_DIM * H_DIM);

    dim3 gg(T_LEN / 16, G_DIM / 64, 1);
    gemm_xg<<<gg, 32, 0, stream>>>(xbf, wfbf, b_ih_f, b_hh_f, xg_f);
    gemm_xg<<<gg, 32, 0, stream>>>(xbf, wbbf, b_ih_b, b_hh_b, xg_b);

    const size_t lds_bytes = (size_t)ROWS_PER_WG * H_DIM * sizeof(__bf16)
                           + (size_t)H_DIM * sizeof(float)
                           + (size_t)ROWS_PER_WG * sizeof(float); // 67840 B
    lstm_kernel<<<2 * NWG, LSTM_THREADS, lds_bytes, stream>>>(
        whfbf, whbbf, xg_f, xg_b, h0, c0, hs_f, hs_b, hbuf, bar);

    feats_kernel<<<(T_LEN * K_TAG + 255) / 256, 256, 0, stream>>>(
        hs_f, hs_b, W_tag, b_tag, feats);

    viterbi_kernel<<<1, 32, 0, stream>>>(feats, trans, out, bptrs);
}
